// FAPM_69552700391749
// MI455X (gfx1250) — matrix operand order repaired and checked by reading, not compiled, not run on hardware
//
#include <hip/hip_runtime.h>
#include <hip/hip_bf16.h>
#include <math.h>

// Problem constants (match reference)
#define NROWS 32768   // embedding rows
#define MROWS 8192    // memory bank rows
#define DDIM  384     // feature dim
#define KNEI  9       // top-k

#define NTILES (MROWS / 16)        // 512 memory tiles of 16 rows
#define NCHUNK (DDIM / 32)         // 12 K-chunks of 32 per tile
#define TILE_BF16_BYTES (NCHUNK * 32 * 16 * 2)   // 12*32 lanes*16 bf16 = 12288 B
#define TILE_STRIDE (TILE_BF16_BYTES + 16 * 4)   // + 16 f32 y_norm header = 12352 B

typedef __attribute__((ext_vector_type(16))) __bf16 v16bf;
typedef __attribute__((ext_vector_type(8)))  float  v8f;

// ---------------------------------------------------------------------------
// Prep: memory_bank f32 -> bf16, pre-swizzled to the CDNA5 WMMA A-operand
// layout (16-bit A 16x32): lane L (half=L/16, row=L%16) holds, for chunk kc,
// element e(0..15): K = (e<8) ? 8*half + e : 16 + 8*half + (e-8).
// Also writes exact f32 y_norm per memory row into a 64B tile header.
// ---------------------------------------------------------------------------
__global__ void __launch_bounds__(256) prep_bank(const float* __restrict__ bank,
                                                 char* __restrict__ ws) {
    const int tb  = blockIdx.x;           // tile index 0..511
    const int tid = threadIdx.x;

    __bf16* outb = (__bf16*)(ws + (size_t)tb * TILE_STRIDE);
    // 6144 bf16 elements per tile, 24 per thread
    for (int idx = tid; idx < NCHUNK * 512; idx += 256) {
        const int kc   = idx >> 9;        // /512
        const int rem  = idx & 511;
        const int lane = rem >> 4;
        const int e    = rem & 15;
        const int r    = e >> 3;
        const int j    = e & 7;
        const int row  = tb * 16 + (lane & 15);
        const int half = lane >> 4;
        const int k    = kc * 32 + r * 16 + half * 8 + j;
        outb[idx] = (__bf16)bank[(size_t)row * DDIM + k];
    }

    // Deterministic y_norm: 16 partials per row, then serial combine.
    __shared__ float partial[16][16];
    {
        const int row = tid >> 4;
        const int p   = tid & 15;
        const float* rp = bank + (size_t)(tb * 16 + row) * DDIM + p * 24;
        float s = 0.f;
        #pragma unroll
        for (int i = 0; i < 24; ++i) { float f = rp[i]; s = fmaf(f, f, s); }
        partial[row][p] = s;
    }
    __syncthreads();
    if (tid < 16) {
        float acc = 0.f;
        #pragma unroll
        for (int i = 0; i < 16; ++i) acc += partial[tid][i];
        *(float*)(ws + (size_t)tb * TILE_STRIDE + TILE_BF16_BYTES + tid * 4) = acc;
    }
}

// ---------------------------------------------------------------------------
// Fused distance + top-9
// ---------------------------------------------------------------------------
__device__ __forceinline__ void cex(float& a, float& b) {
    float lo = fminf(a, b);
    b = fmaxf(a, b);
    a = lo;
}
__device__ __forceinline__ void ins9(float (&t)[9], float v) {
    if (v < t[8]) {
        t[8] = v;
        cex(t[7], t[8]); cex(t[6], t[7]); cex(t[5], t[6]); cex(t[4], t[5]);
        cex(t[3], t[4]); cex(t[2], t[3]); cex(t[1], t[2]); cex(t[0], t[1]);
    }
}

__global__ void __launch_bounds__(256) topk_dist(const float* __restrict__ emb,
                                                 const char* __restrict__ ws,
                                                 float* __restrict__ out) {
    const int tid  = threadIdx.x;
    const int wave = tid >> 5;
    const int lane = tid & 31;
    const int col  = lane & 15;          // C column = embedding row within wave
    const int half = lane >> 4;

    const int erow = blockIdx.x * 128 + wave * 16 + col;

    // --- Load B operand (16 embedding rows, bf16) into registers; fold x_norm.
    // B layout (16-bit 32x16): lane col=L%16; element e: K = (e<8) ? 8*half + e : 16 + 8*half + (e-8).
    v16bf B[NCHUNK];
    float part = 0.f;
    {
        const float* rp = emb + (size_t)erow * DDIM;
        #pragma unroll
        for (int kc = 0; kc < NCHUNK; ++kc) {
            const float4* r4 = (const float4*)(rp + kc * 32 + half * 8);
            float4 f0 = r4[0], f1 = r4[1], f2 = r4[4], f3 = r4[5];
            part += f0.x*f0.x + f0.y*f0.y + f0.z*f0.z + f0.w*f0.w;
            part += f1.x*f1.x + f1.y*f1.y + f1.z*f1.z + f1.w*f1.w;
            part += f2.x*f2.x + f2.y*f2.y + f2.z*f2.z + f2.w*f2.w;
            part += f3.x*f3.x + f3.y*f3.y + f3.z*f3.z + f3.w*f3.w;
            v16bf b;
            b[0]=(__bf16)f0.x; b[1]=(__bf16)f0.y; b[2]=(__bf16)f0.z; b[3]=(__bf16)f0.w;
            b[4]=(__bf16)f1.x; b[5]=(__bf16)f1.y; b[6]=(__bf16)f1.z; b[7]=(__bf16)f1.w;
            b[8]=(__bf16)f2.x; b[9]=(__bf16)f2.y; b[10]=(__bf16)f2.z; b[11]=(__bf16)f2.w;
            b[12]=(__bf16)f3.x; b[13]=(__bf16)f3.y; b[14]=(__bf16)f3.z; b[15]=(__bf16)f3.w;
            B[kc] = b;
        }
    }
    const float xn = part + __shfl_xor(part, 16, 32);   // full-row ||x||^2

    float t9[9];
    #pragma unroll
    for (int j = 0; j < 9; ++j) t9[j] = 3.0e38f;

    // --- Stream all memory tiles (A operands pre-swizzled in ws, L2-resident).
    const char* tileBase = ws + (size_t)lane * 32;
    for (int t = 0; t < NTILES; ++t) {
        const v16bf* aptr = (const v16bf*)tileBase;     // +lane*32 already folded
        const float4* hp  = (const float4*)(ws + (size_t)t * TILE_STRIDE
                                            + TILE_BF16_BYTES + half * 32);
        float4 y0 = hp[0], y1 = hp[1];                  // y_norm rows 8*half..+7

        v8f c = {};
        #pragma unroll
        for (int kc = 0; kc < NCHUNK; ++kc) {
            v16bf a = aptr[kc * 32];                    // byte offset kc*1024
            c = __builtin_amdgcn_wmma_f32_16x16x32_bf16(
                    false, a, false, B[kc], (short)0, c, false, false);
        }

        const float yv[8] = { y0.x, y0.y, y0.z, y0.w, y1.x, y1.y, y1.z, y1.w };
        #pragma unroll
        for (int v = 0; v < 8; ++v) {
            float d2 = fmaf(-2.0f, c[v], xn + yv[v]);
            ins9(t9, d2);
        }
        tileBase += TILE_STRIDE;
    }

    // --- Merge the two half-lanes covering the same embedding row.
    float pv[9];
    #pragma unroll
    for (int j = 0; j < 9; ++j) pv[j] = __shfl_xor(t9[j], 16, 32);
    #pragma unroll
    for (int j = 0; j < 9; ++j) ins9(t9, pv[j]);

    if (lane < 16) {
        float* o = out + (size_t)erow * KNEI;
        #pragma unroll
        for (int j = 0; j < 9; ++j) o[j] = sqrtf(fmaxf(t9[j], 0.f));
    }
}

// ---------------------------------------------------------------------------
extern "C" void kernel_launch(void* const* d_in, const int* in_sizes, int n_in,
                              void* d_out, int out_size, void* d_ws, size_t ws_size,
                              hipStream_t stream) {
    (void)in_sizes; (void)n_in; (void)out_size; (void)ws_size;
    const float* emb  = (const float*)d_in[0];
    const float* bank = (const float*)d_in[1];
    float* out = (float*)d_out;
    char*  ws  = (char*)d_ws;   // needs 512 * 12352 = 6.3 MB

    prep_bank<<<NTILES, 256, 0, stream>>>(bank, ws);
    topk_dist<<<NROWS / 128, 256, 0, stream>>>(emb, ws, out);
}
